// LSTMModel_12395275616962
// MI455X (gfx1250) — matrix operand order repaired and checked by reading, not compiled, not run on hardware
//
#include <hip/hip_runtime.h>

// ---------------------------------------------------------------------------
// Bidirectional 2-layer LSTM for MI455X (gfx1250, wave32, WMMA).
//   B=128, T=512, V=100000, E=100, H=128, C=3
// Strategy:
//   1) embed+pad -> xs bf16 (T*B x 128)
//   2) device-side weight swizzle into per-lane WMMA B-fragment layout
//      (consecutive N-tiles +1024B, K-chunks +32768B -> immediate-offset loads)
//   3) gx = X @ W_ih^T + b  as bf16 WMMA GEMMs (fp32 out, nontemporal stream)
//      with all 8 B-fragments loaded per K-chunk before the WMMA chain (ILP)
//   4) recurrent scan: 16-batch-row chunks x 2 dirs; w_hh B-fragments resident
//      in VGPRs for all 512 steps; h in LDS (bank-conflict-padded); next-step
//      gx tile prefetched (global_prefetch) to hide HBM latency; gates via
//      v_wmma_f32_16x16x32_bf16 with fp32 accum; c-state in registers.
//   5) tiny FC -> d_out (128x3 fp32)
// Workspace requirement: ~320 MB.
// ---------------------------------------------------------------------------

typedef __attribute__((ext_vector_type(16))) __bf16 v16bf;
typedef __attribute__((ext_vector_type(8)))  float  v8f;

union BF16Frag { uint4 u[2]; v16bf v; };
union HPack    { unsigned short s[8]; uint4 u; };

static constexpr int BB = 128;   // batch
static constexpr int TT = 512;   // time
static constexpr int EE = 100;   // embed dim (padded to 128)
static constexpr int GG = 512;   // 4*H gates

__device__ __forceinline__ unsigned short f2bf(float f) {
  unsigned u = __float_as_uint(f);
  u += 0x7FFFu + ((u >> 16) & 1u);           // round-to-nearest-even
  return (unsigned short)(u >> 16);
}
__device__ __forceinline__ float sigf(float x)  { return 1.f / (1.f + __expf(-x)); }
__device__ __forceinline__ float tanhf_(float x){ return 2.f / (1.f + __expf(-2.f*x)) - 1.f; }

__device__ __forceinline__ v8f zero8() {
  v8f z;
#pragma unroll
  for (int i = 0; i < 8; ++i) z[i] = 0.f;
  return z;
}

// ---------------------------------------------------------------------------
// 1) Embedding gather: xs[(t*B+b)*128 + col] = bf16(emb[x[b,t], col]) (pad->0)
// ---------------------------------------------------------------------------
__global__ __launch_bounds__(256) void embed_pad(const int* __restrict__ x,
                                                 const float* __restrict__ emb,
                                                 unsigned short* __restrict__ xs) {
  const long long gid = (long long)blockIdx.x * 256 + threadIdx.x;
  const int col = (int)(gid & 127);
  const long long tk = gid >> 7;              // t*B + b
  const int t = (int)(tk >> 7);               // B == 128
  const int b = (int)(tk & 127);
  const int token = x[(size_t)b * TT + t];
  float v = (col < EE) ? emb[(size_t)token * EE + col] : 0.f;
  xs[(size_t)tk * 128 + col] = f2bf(v);
}

// ---------------------------------------------------------------------------
// 2) Swizzle weight (N=512 rows x din cols, row-major fp32) into WMMA
//    B-operand tiles: out[((kc*32+nt)*32+lane)*16+e] = W[n, k]
//    with n = nt*16 + (lane&15), k = kc*32 + ((e>=8)?16:0) + ((lane>=16)?8:0) + (e&7), 0 if k>=din
//    Assumed bf16 B layout (32x16 K-major per-lane): lane<16 holds K {0..7,16..23},
//    lane>=16 holds K {8..15,24..31} of the 32-chunk; N = lane%16.
// ---------------------------------------------------------------------------
__global__ __launch_bounds__(256) void swizzle_w(const float* __restrict__ w,
                                                 unsigned short* __restrict__ out,
                                                 int din, int KC) {
  const int gid = blockIdx.x * 256 + threadIdx.x;   // KC*16384 total
  const int e    = gid & 15;
  const int lane = (gid >> 4) & 31;
  const int nt   = (gid >> 9) & 31;
  const int kc   = gid >> 14;
  if (kc >= KC) return;
  const int k = kc * 32 + ((e >> 3) << 4) + ((lane >> 4) << 3) + (e & 7);
  const int n = nt * 16 + (lane & 15);
  out[gid] = (k < din) ? f2bf(w[(size_t)n * din + k]) : (unsigned short)0;
}

__global__ void bias_sum(const float* __restrict__ bi, const float* __restrict__ bh,
                         float* __restrict__ o) {
  const int i = blockIdx.x * 256 + threadIdx.x;
  if (i < GG) o[i] = bi[i] + bh[i];
}

// ---------------------------------------------------------------------------
// 3) gx GEMM: gx(M x 512) = X(M x Kd, bf16) @ Wsw + bias.  Block = 128x128 tile,
//    8 waves; wave w computes rows [16w,16w+16) x 8 N-tiles via WMMA chains.
//    Per K-chunk: one clause of 16 b128 loads (all 8 B-frags), then 8 WMMAs.
// ---------------------------------------------------------------------------
__global__ __launch_bounds__(256) void gemm_gx(const unsigned short* __restrict__ X,
                                               const unsigned short* __restrict__ Wsw,
                                               const float* __restrict__ bias,
                                               float* __restrict__ gx, int Kd) {
  const int tid = threadIdx.x, wv = tid >> 5, lane = tid & 31;
  const int l15 = lane & 15, hi = lane >> 4;
  const int mrow = blockIdx.x * 128 + wv * 16 + l15;   // A row for this lane
  const int KC = Kd >> 5;
  v8f acc[8];
#pragma unroll
  for (int nt = 0; nt < 8; ++nt) acc[nt] = zero8();

  // Single base pointers; all inner-loop offsets become immediates.
  const unsigned short* ap = X + (size_t)mrow * Kd + hi * 8;
  const unsigned short* bp =
      Wsw + ((size_t)(blockIdx.y * 8) * 32 + lane) * 16;   // +nt*512, +kc*16384

  for (int kc = 0; kc < KC; ++kc) {
    BF16Frag a;   // A 16x32 bf16: lane<16 -> K {0..7,16..23}; lane>=16 -> +8
    a.u[0] = *reinterpret_cast<const uint4*>(ap);
    a.u[1] = *reinterpret_cast<const uint4*>(ap + 16);
    BF16Frag bf[8];
#pragma unroll
    for (int nt = 0; nt < 8; ++nt) {       // 16 b128 loads, clause-able
      bf[nt].u[0] = *reinterpret_cast<const uint4*>(bp + nt * 512);
      bf[nt].u[1] = *reinterpret_cast<const uint4*>(bp + nt * 512 + 8);
    }
#pragma unroll
    for (int nt = 0; nt < 8; ++nt)
      acc[nt] = __builtin_amdgcn_wmma_f32_16x16x32_bf16(
          false, a.v, false, bf[nt].v, (short)0, acc[nt], false, false);
    ap += 32;
    bp += 16384;                           // 32 N-tiles * 512 ushorts
  }
#pragma unroll
  for (int nt = 0; nt < 8; ++nt) {
    const int ncol = blockIdx.y * 128 + nt * 16 + l15;
    const float bv = bias[ncol];
#pragma unroll
    for (int r = 0; r < 8; ++r) {
      const int mg = blockIdx.x * 128 + wv * 16 + r + hi * 8;   // C layout
      __builtin_nontemporal_store(acc[nt][r] + bv, &gx[(size_t)mg * GG + ncol]);
    }
  }
}

// ---------------------------------------------------------------------------
// 4) Recurrent scan. grid = (8 batch-chunks, 2 dirs), 256 threads (8 waves).
//    Each block: 16 batch rows, all 512 steps. w_hh fragments register-resident.
// ---------------------------------------------------------------------------
template <int LAYER>
__global__ __launch_bounds__(256, 1) void lstm_scan(
    const float* __restrict__ gxF, const float* __restrict__ gxR,
    const unsigned short* __restrict__ whhF, const unsigned short* __restrict__ whhR,
    unsigned short* __restrict__ x1out, float* __restrict__ hlast) {
  __shared__ unsigned short hS[16 * 136];   // h bf16, row stride 136 (16B pad)
  __shared__ float gS[16 * 516];            // gate staging, row stride 516

  const int tid = threadIdx.x, wv = tid >> 5, lane = tid & 31;
  const int l15 = lane & 15, hi = lane >> 4;
  const int dir = blockIdx.y;
  const int b0  = blockIdx.x * 16;
  const float* gx = dir ? gxR : gxF;
  const unsigned short* whh = dir ? whhR : whhF;

  // Hold all 16 w_hh B-fragments (4 N-tiles x 4 K-chunks) in VGPRs for the
  // whole scan: 128 VGPRs/lane on wave32 — removes weight traffic entirely.
  v16bf Bf[16];
  {
    const unsigned short* bp = whh + ((size_t)(wv * 4) * 32 + lane) * 16;
#pragma unroll
    for (int kc = 0; kc < 4; ++kc)
#pragma unroll
      for (int ntl = 0; ntl < 4; ++ntl) {
        BF16Frag b;
        b.u[0] = *reinterpret_cast<const uint4*>(bp + kc * 16384 + ntl * 512);
        b.u[1] = *reinterpret_cast<const uint4*>(bp + kc * 16384 + ntl * 512 + 8);
        Bf[kc * 4 + ntl] = b.v;
      }
  }

  for (int i = tid; i < 16 * 136; i += 256) hS[i] = 0;
  float cst[8];
#pragma unroll
  for (int q = 0; q < 8; ++q) cst[q] = 0.f;
  const int em = tid >> 4, jb = (tid & 15) * 8;   // elementwise ownership
  __syncthreads();

  for (int s = 0; s < TT; ++s) {
    const int t = dir ? (TT - 1 - s) : s;

    // Prefetch next step's 32KB gx tile (256 threads x 128B -> full tile);
    // hides HBM latency behind this step's WMMA + transcendental work.
    if (s + 1 < TT) {
      const int tn = dir ? (TT - 2 - s) : (s + 1);
      const float* pf = &gx[((size_t)tn * BB + b0 + (tid >> 4)) * GG + (tid & 15) * 32];
      __builtin_prefetch(pf, 0, 1);
    }

    // A fragments (current h) from LDS: two b128 ds loads per K-chunk.
    v16bf Af[4];
#pragma unroll
    for (int kc = 0; kc < 4; ++kc) {
      BF16Frag a;
      const unsigned short* apx = &hS[l15 * 136 + kc * 32 + hi * 8];
      a.u[0] = *reinterpret_cast<const uint4*>(apx);
      a.u[1] = *reinterpret_cast<const uint4*>(apx + 16);
      Af[kc] = a.v;
    }

#pragma unroll
    for (int ntl = 0; ntl < 4; ++ntl) {
      v8f acc = zero8();
#pragma unroll
      for (int kc = 0; kc < 4; ++kc)
        acc = __builtin_amdgcn_wmma_f32_16x16x32_bf16(
            false, Af[kc], false, Bf[kc * 4 + ntl], (short)0, acc, false, false);
      const int ncol = wv * 64 + ntl * 16 + l15;
      const size_t g0 = ((size_t)t * BB + b0) * GG + ncol;
#pragma unroll
      for (int r = 0; r < 8; ++r) {
        const int m = r + hi * 8;   // C layout row
        gS[m * 516 + ncol] = acc[r] + __builtin_nontemporal_load(&gx[g0 + (size_t)m * GG]);
      }
    }
    __syncthreads();

    // Gate update: thread owns (row em, cols jb..jb+7); c stays in registers.
    const float* gr = &gS[em * 516];
    float zi[8], zf[8], zg[8], zo[8], hv[8];
    *(float4*)(zi)     = *(const float4*)&gr[jb];       *(float4*)(zi + 4) = *(const float4*)&gr[jb + 4];
    *(float4*)(zf)     = *(const float4*)&gr[128 + jb]; *(float4*)(zf + 4) = *(const float4*)&gr[128 + jb + 4];
    *(float4*)(zg)     = *(const float4*)&gr[256 + jb]; *(float4*)(zg + 4) = *(const float4*)&gr[256 + jb + 4];
    *(float4*)(zo)     = *(const float4*)&gr[384 + jb]; *(float4*)(zo + 4) = *(const float4*)&gr[384 + jb + 4];
    HPack hp;
#pragma unroll
    for (int q = 0; q < 8; ++q) {
      const float iv = sigf(zi[q]), fv = sigf(zf[q]);
      const float gv = tanhf_(zg[q]), ov = sigf(zo[q]);
      cst[q] = fv * cst[q] + iv * gv;
      hv[q] = ov * tanhf_(cst[q]);
      hp.s[q] = f2bf(hv[q]);
    }
    *reinterpret_cast<uint4*>(&hS[em * 136 + jb]) = hp.u;
    if (LAYER == 0) {
      // x1[t, b, dir*128 + j] in bf16 for the layer-1 input GEMM
      *reinterpret_cast<uint4*>(
          x1out + ((size_t)t * BB + b0 + em) * 256 + dir * 128 + jb) = hp.u;
    } else if (s == TT - 1) {
      float* hp2 = hlast + (size_t)(b0 + em) * 256 + dir * 128 + jb;
      *(float4*)hp2       = *(float4*)hv;
      *(float4*)(hp2 + 4) = *(float4*)(hv + 4);
    }
    __syncthreads();
  }
}

// ---------------------------------------------------------------------------
// 5) Final FC: out(128x3) = hcat(128x256) @ fc_w^T + fc_b
// ---------------------------------------------------------------------------
__global__ __launch_bounds__(128) void fc_k(const float* __restrict__ hcat,
                                            const float* __restrict__ w,
                                            const float* __restrict__ b,
                                            float* __restrict__ out) {
  const int bi = threadIdx.x;
#pragma unroll
  for (int c = 0; c < 3; ++c) {
    float a = b[c];
    for (int k = 0; k < 256; ++k) a += hcat[(size_t)bi * 256 + k] * w[(size_t)c * 256 + k];
    out[bi * 3 + c] = a;
  }
}

// ---------------------------------------------------------------------------
extern "C" void kernel_launch(void* const* d_in, const int* in_sizes, int n_in,
                              void* d_out, int out_size, void* d_ws, size_t ws_size,
                              hipStream_t stream) {
  (void)in_sizes; (void)n_in; (void)out_size; (void)ws_size;
  const int*   x        = (const int*)  d_in[0];
  const float* emb      = (const float*)d_in[1];
  const float* w_ih_l0  = (const float*)d_in[2];
  const float* w_hh_l0  = (const float*)d_in[3];
  const float* b_ih_l0  = (const float*)d_in[4];
  const float* b_hh_l0  = (const float*)d_in[5];
  const float* w_ih_l0r = (const float*)d_in[6];
  const float* w_hh_l0r = (const float*)d_in[7];
  const float* b_ih_l0r = (const float*)d_in[8];
  const float* b_hh_l0r = (const float*)d_in[9];
  const float* w_ih_l1  = (const float*)d_in[10];
  const float* w_hh_l1  = (const float*)d_in[11];
  const float* b_ih_l1  = (const float*)d_in[12];
  const float* b_hh_l1  = (const float*)d_in[13];
  const float* w_ih_l1r = (const float*)d_in[14];
  const float* w_hh_l1r = (const float*)d_in[15];
  const float* b_ih_l1r = (const float*)d_in[16];
  const float* b_hh_l1r = (const float*)d_in[17];
  const float* fc_w     = (const float*)d_in[18];
  const float* fc_b     = (const float*)d_in[19];

  char* wp = (char*)d_ws;
  auto alloc = [&](size_t n) -> void* {
    void* p = wp; wp += (n + 255) & ~(size_t)255; return p;
  };
  const size_t TB = (size_t)TT * BB;                         // 65536
  unsigned short* xs  = (unsigned short*)alloc(TB * 128 * 2);  // 16.8 MB
  unsigned short* x1  = (unsigned short*)alloc(TB * 256 * 2);  // 33.6 MB
  float* gxA = (float*)alloc(TB * 512 * 4);                    // 134 MB (reused)
  float* gxB = (float*)alloc(TB * 512 * 4);                    // 134 MB (reused)
  unsigned short* wih0f = (unsigned short*)alloc(4 * 16384 * 2);
  unsigned short* wih0r = (unsigned short*)alloc(4 * 16384 * 2);
  unsigned short* whh0f = (unsigned short*)alloc(4 * 16384 * 2);
  unsigned short* whh0r = (unsigned short*)alloc(4 * 16384 * 2);
  unsigned short* wih1f = (unsigned short*)alloc(8 * 16384 * 2);
  unsigned short* wih1r = (unsigned short*)alloc(8 * 16384 * 2);
  unsigned short* whh1f = (unsigned short*)alloc(4 * 16384 * 2);
  unsigned short* whh1r = (unsigned short*)alloc(4 * 16384 * 2);
  float* bs0f = (float*)alloc(GG * 4);
  float* bs0r = (float*)alloc(GG * 4);
  float* bs1f = (float*)alloc(GG * 4);
  float* bs1r = (float*)alloc(GG * 4);
  float* hcat = (float*)alloc(128 * 256 * 4);

  // 1) embedding gather + pad
  embed_pad<<<(int)(TB * 128 / 256), 256, 0, stream>>>(x, emb, xs);

  // 2) weight swizzles + bias sums
  swizzle_w<<<4 * 64, 256, 0, stream>>>(w_ih_l0,  wih0f, EE,  4);
  swizzle_w<<<4 * 64, 256, 0, stream>>>(w_ih_l0r, wih0r, EE,  4);
  swizzle_w<<<4 * 64, 256, 0, stream>>>(w_hh_l0,  whh0f, 128, 4);
  swizzle_w<<<4 * 64, 256, 0, stream>>>(w_hh_l0r, whh0r, 128, 4);
  swizzle_w<<<8 * 64, 256, 0, stream>>>(w_ih_l1,  wih1f, 256, 8);
  swizzle_w<<<8 * 64, 256, 0, stream>>>(w_ih_l1r, wih1r, 256, 8);
  swizzle_w<<<4 * 64, 256, 0, stream>>>(w_hh_l1,  whh1f, 128, 4);
  swizzle_w<<<4 * 64, 256, 0, stream>>>(w_hh_l1r, whh1r, 128, 4);
  bias_sum<<<2, 256, 0, stream>>>(b_ih_l0,  b_hh_l0,  bs0f);
  bias_sum<<<2, 256, 0, stream>>>(b_ih_l0r, b_hh_l0r, bs0r);
  bias_sum<<<2, 256, 0, stream>>>(b_ih_l1,  b_hh_l1,  bs1f);
  bias_sum<<<2, 256, 0, stream>>>(b_ih_l1r, b_hh_l1r, bs1r);

  // 3) layer-0 input projections (WMMA GEMM), then scan
  gemm_gx<<<dim3(512, 4), 256, 0, stream>>>(xs, wih0f, bs0f, gxA, 128);
  gemm_gx<<<dim3(512, 4), 256, 0, stream>>>(xs, wih0r, bs0r, gxB, 128);
  lstm_scan<0><<<dim3(8, 2), 256, 0, stream>>>(gxA, gxB, whh0f, whh0r, x1, nullptr);

  // 4) layer-1 projections (reuse gx buffers), then scan producing h_last
  gemm_gx<<<dim3(512, 4), 256, 0, stream>>>(x1, wih1f, bs1f, gxA, 256);
  gemm_gx<<<dim3(512, 4), 256, 0, stream>>>(x1, wih1r, bs1r, gxB, 256);
  lstm_scan<1><<<dim3(8, 2), 256, 0, stream>>>(gxA, gxB, whh1f, whh1r, nullptr, hcat);

  // 5) FC head
  fc_k<<<1, 128, 0, stream>>>(hcat, fc_w, fc_b, (float*)d_out);
}
